// NeptuneMoEModel_29953101923026
// MI455X (gfx1250) — compile-verified, matrix operand order checked by reading, not run on hardware
//
#include <hip/hip_runtime.h>
#include <hip/hip_bf16.h>
#include <math.h>

// ---------------------------------------------------------------------------
// NeptuneMoE: 6 point-MLPs (9->256->256) + segment-mean pool + tiny heads.
// Compute-bound (~214 GFLOP vs ~10MB traffic) -> v_wmma_f32_16x16x32_f16.
// GELU via branch-free v_exp_f32/v_rcp_f32; next-net weights prefetched.
// ---------------------------------------------------------------------------

typedef __attribute__((ext_vector_type(16))) _Float16 v16h;
typedef __attribute__((ext_vector_type(8)))  _Float16 h8;
typedef __attribute__((ext_vector_type(8)))  float    v8f;

#define NPTS   262144
#define NB     1024
#define NH     256
#define NDIN   9
#define NNETS  6
#define RPB    128      // rows (points) per block
#define NWAVES 8

// LDS layout (halfs):
//   xa  : 128*32            input tile, K padded 9->32           (8 KB)
//   w1f : 16*32*16          W1 in B-fragment layout              (16 KB)
//   w2f : 16*8*32*16        W2 in B-fragment layout              (128 KB)
//   h1  : 128*256           layer-1 activations (f16)            (64 KB)
//   b1s/b2s : 256 f32 each                                       (2 KB)
#define LDS_XA   (RPB*32)
#define LDS_W1F  (16*32*16)
#define LDS_W2F  (16*8*32*16)
#define LDS_H1   (RPB*NH)
#define LDS_BYTES ((LDS_XA + LDS_W1F + LDS_W2F + LDS_H1)*2 + 2*NH*4)

// Branch-free tanh-approx GELU:
//   gelu(x) = x * sigmoid( 2*sqrt(2/pi) * (x + 0.044715 x^3) )
// sigmoid(z) = 1 / (1 + exp2(-z*log2e)), via v_exp_f32 + v_rcp_f32.
__device__ __forceinline__ float gelu_f(float x) {
    float u = x * x;
    float z = x * __builtin_fmaf(0.07135481283f, u, 1.5957691216f);
    float e = __builtin_amdgcn_exp2f(-1.4426950408889634f * z);
    return x * __builtin_amdgcn_rcpf(1.0f + e);
}

__device__ __forceinline__ float sigmoid_f(float x) {
    float e = __builtin_amdgcn_exp2f(-1.4426950408889634f * x);
    return __builtin_amdgcn_rcpf(1.0f + e);
}

__device__ __forceinline__ float exp_f(float x) {
    return __builtin_amdgcn_exp2f(1.4426950408889634f * x);
}

__global__ void neptune_init_kernel(float* p, int n) {
    int i = blockIdx.x * blockDim.x + threadIdx.x;
    if (i < n) p[i] = 0.0f;
}

__global__ void neptune_count_kernel(const int* __restrict__ bids, float* __restrict__ cnt) {
    int i = blockIdx.x * blockDim.x + threadIdx.x;
    atomicAdd(&cnt[bids[i]], 1.0f);
}

// ---------------------------------------------------------------------------
// Fused per-point MLP (both layers, all 6 nets) + pooled segment sums
// ---------------------------------------------------------------------------
__global__ __launch_bounds__(256, 1)
void neptune_mlp_kernel(const float* __restrict__ coords,
                        const float* __restrict__ feats,
                        const int*   __restrict__ bids,
                        const float* __restrict__ rW1, const float* __restrict__ rb1,
                        const float* __restrict__ rW2, const float* __restrict__ rb2,
                        const float* __restrict__ eW1, const float* __restrict__ eb1,
                        const float* __restrict__ eW2, const float* __restrict__ eb2,
                        const float* __restrict__ dW1, const float* __restrict__ db1,
                        const float* __restrict__ dW2, const float* __restrict__ db2,
                        float* __restrict__ sums) {
    extern __shared__ __align__(32) char smem[];
    _Float16* xa  = (_Float16*)smem;
    _Float16* w1f = xa  + LDS_XA;
    _Float16* w2f = w1f + LDS_W1F;
    _Float16* h1  = w2f + LDS_W2F;
    float*    b1s = (float*)(h1 + LDS_H1);
    float*    b2s = b1s + NH;

    const int tid  = threadIdx.x;
    const int lane = tid & 31;
    const int wave = tid >> 5;
    const int row0 = blockIdx.x * RPB;     // first point of this block
    const int wr0  = wave * 16;            // wave's row tile within block

    const int lane16 = lane & 15;
    const int rowsel = (lane < 16) ? 0 : 8;  // row offset for C frags, K base for A/B frags

    // --- stage input tile [128 x 32] (K padded with zeros) ---
    for (int idx = tid; idx < RPB * 32; idx += 256) {
        int row = idx >> 5, k = idx & 31;
        int g = row0 + row;
        float v = 0.0f;
        if (k < 3)       v = coords[g * 3 + k];
        else if (k < 9)  v = feats[g * 6 + (k - 3)];
        xa[idx] = (_Float16)v;
    }

    // per-lane batch ids for this wave's rows (sorted input -> cheap uniform test)
    int mybid[8];
    #pragma unroll
    for (int r = 0; r < 8; ++r)
        mybid[r] = bids[row0 + wr0 + rowsel + r];
    const int b_first  = bids[row0 + wr0];
    const int b_last   = bids[row0 + wr0 + 15];
    const bool uniform = (b_first == b_last);

    const int myrow = wr0 + lane16;

    for (int net = 0; net < NNETS; ++net) {
        // select this net's weights
        const float *W1, *B1, *W2, *B2;
        if (net == 0)      { W1 = rW1;                   B1 = rb1;                W2 = rW2;                    B2 = rb2; }
        else if (net < 3)  { int e = net - 1; W1 = eW1 + e*NDIN*NH; B1 = eb1 + e*NH; W2 = eW2 + (size_t)e*NH*NH; B2 = eb2 + e*NH; }
        else               { int e = net - 3; W1 = dW1 + e*NDIN*NH; B1 = db1 + e*NH; W2 = dW2 + (size_t)e*NH*NH; B2 = db2 + e*NH; }

        __syncthreads();   // all waves done reading previous net's weight tiles

        // --- stage W1 (9x256 -> padded 32x256) directly in B-fragment layout ---
        for (int idx = tid; idx < LDS_W1F; idx += 256) {
            int elem = idx & 15;
            int ln   = (idx >> 4) & 31;
            int nt   = idx >> 9;
            int cl   = ln & 15;
            int base = (ln < 16) ? 0 : 8;
            int K    = (elem < 8) ? (base + elem) : (16 + base + (elem - 8));
            float v  = (K < NDIN) ? W1[K * NH + nt * 16 + cl] : 0.0f;
            w1f[idx] = (_Float16)v;
        }
        // --- stage W2 (256x256) swizzled into B-fragment layout, b128 loads ---
        // 4 consecutive columns share nt/ks/elem/hi (c0 is 4-aligned), dest
        // stride is 16 halfs per column.
        for (int idx4 = tid; idx4 < (NH * NH) / 4; idx4 += 256) {
            int base = idx4 * 4;
            int K = base >> 8, c0 = base & 255;
            const float4 v4 = *(const float4*)(W2 + base);
            int nt = c0 >> 4;
            int ks = K >> 5, kk = K & 31;
            int hi   = (kk >> 3) & 1;
            int elem = kk - hi * 8 - ((kk & 16) ? 8 : 0);
            int dbase = (((nt * 8 + ks) * 32) + ((c0 & 15) + hi * 16)) * 16 + elem;
            w2f[dbase]      = (_Float16)v4.x;
            w2f[dbase + 16] = (_Float16)v4.y;
            w2f[dbase + 32] = (_Float16)v4.z;
            w2f[dbase + 48] = (_Float16)v4.w;
        }
        if (tid < NH) { b1s[tid] = B1[tid]; b2s[tid] = B2[tid]; }
        __syncthreads();

        // --- prefetch next net's W2 into cache while this net computes ---
        if (net + 1 < NNETS) {
            const float* nW2;
            int nn = net + 1;
            if (nn < 3) nW2 = eW2 + (size_t)(nn - 1) * NH * NH;
            else        nW2 = dW2 + (size_t)(nn - 3) * NH * NH;
            for (int off = tid * 32; off < NH * NH; off += 256 * 32)
                __builtin_prefetch(nW2 + off, 0, 0);
        }

        // --- layer 1: [16x32] x [32x256] via 16 WMMAs ---
        {
            h8 lo = *(const h8*)(xa + myrow * 32 + rowsel);
            h8 hi = *(const h8*)(xa + myrow * 32 + 16 + rowsel);
            v16h a1 = __builtin_shufflevector(lo, hi, 0,1,2,3,4,5,6,7,8,9,10,11,12,13,14,15);
            #pragma unroll
            for (int nt = 0; nt < 16; ++nt) {
                v16h bfr = *(const v16h*)(w1f + (nt * 32 + lane) * 16);
                v8f c = {0.f,0.f,0.f,0.f,0.f,0.f,0.f,0.f};
                c = __builtin_amdgcn_wmma_f32_16x16x32_f16(false, a1, false, bfr,
                                                           (short)0, c, false, false);
                float bb = b1s[nt * 16 + lane16];
                #pragma unroll
                for (int r = 0; r < 8; ++r) {
                    float v = c[r] + bb;
                    h1[(wr0 + r + rowsel) * NH + nt * 16 + lane16] = (_Float16)gelu_f(v);
                }
            }
        }

        // --- layer 2: [16x256] x [256x256] via 16x8 WMMAs; fused pooling ---
        {
            v16h a2[8];
            #pragma unroll
            for (int k = 0; k < 8; ++k) {
                h8 lo = *(const h8*)(h1 + myrow * NH + k * 32 + rowsel);
                h8 hi = *(const h8*)(h1 + myrow * NH + k * 32 + 16 + rowsel);
                a2[k] = __builtin_shufflevector(lo, hi, 0,1,2,3,4,5,6,7,8,9,10,11,12,13,14,15);
            }
            float* srow = sums + (size_t)net * NB * NH;
            #pragma unroll
            for (int nt = 0; nt < 16; ++nt) {
                v8f acc = {0.f,0.f,0.f,0.f,0.f,0.f,0.f,0.f};
                #pragma unroll
                for (int k = 0; k < 8; ++k) {
                    v16h bfr = *(const v16h*)(w2f + ((nt * 8 + k) * 32 + lane) * 16);
                    acc = __builtin_amdgcn_wmma_f32_16x16x32_f16(false, a2[k], false, bfr,
                                                                 (short)0, acc, false, false);
                }
                float bb = b2s[nt * 16 + lane16];
                if (uniform) {
                    // all 16 rows of this tile belong to one event: reduce then 1 atomic/col
                    float s = 0.0f;
                    #pragma unroll
                    for (int r = 0; r < 8; ++r) s += gelu_f(acc[r] + bb);
                    s += __shfl_xor(s, 16, 32);
                    if (lane < 16)
                        atomicAdd(srow + (size_t)b_first * NH + nt * 16 + lane, s);
                } else {
                    #pragma unroll
                    for (int r = 0; r < 8; ++r)
                        atomicAdd(srow + (size_t)mybid[r] * NH + nt * 16 + lane16,
                                  gelu_f(acc[r] + bb));
                }
            }
        }
    }
}

// ---------------------------------------------------------------------------
// Per-event heads: pooled @ Wh + bh, softmax routing, gating, final [B,11]
// ---------------------------------------------------------------------------
__global__ __launch_bounds__(256)
void neptune_head_kernel(const float* __restrict__ sums, const float* __restrict__ cnt,
                         const float* __restrict__ rWh, const float* __restrict__ rbh,
                         const float* __restrict__ eWh, const float* __restrict__ ebh,
                         const float* __restrict__ dWh, const float* __restrict__ dbh,
                         float* __restrict__ out) {
    const int b = blockIdx.x;
    const int t = threadIdx.x;
    __shared__ float red[8];
    __shared__ float res[19];

    const float ic = 1.0f / fmaxf(cnt[b], 1.0f);
    const float pr  = sums[(size_t)0 * NB * NH + (size_t)b * NH + t] * ic;
    const float pe0 = sums[(size_t)1 * NB * NH + (size_t)b * NH + t] * ic;
    const float pe1 = sums[(size_t)2 * NB * NH + (size_t)b * NH + t] * ic;
    const float pd0 = sums[(size_t)3 * NB * NH + (size_t)b * NH + t] * ic;
    const float pd1 = sums[(size_t)4 * NB * NH + (size_t)b * NH + t] * ic;
    const float pd2 = sums[(size_t)5 * NB * NH + (size_t)b * NH + t] * ic;

    float part[19];
    #pragma unroll
    for (int j = 0; j < 6; ++j) part[j] = pr * rWh[t * 6 + j];
    #pragma unroll
    for (int j = 0; j < 2; ++j) part[6 + j] = pe0 * eWh[t * 2 + j];
    #pragma unroll
    for (int j = 0; j < 2; ++j) part[8 + j] = pe1 * eWh[NH * 2 + t * 2 + j];
    #pragma unroll
    for (int j = 0; j < 3; ++j) part[10 + j] = pd0 * dWh[t * 3 + j];
    #pragma unroll
    for (int j = 0; j < 3; ++j) part[13 + j] = pd1 * dWh[NH * 3 + t * 3 + j];
    #pragma unroll
    for (int j = 0; j < 3; ++j) part[16 + j] = pd2 * dWh[2 * NH * 3 + t * 3 + j];

    for (int o = 0; o < 19; ++o) {
        float v = part[o];
        #pragma unroll
        for (int off = 16; off > 0; off >>= 1) v += __shfl_xor(v, off, 32);
        if ((t & 31) == 0) red[t >> 5] = v;
        __syncthreads();
        if (t == 0) {
            float s = 0.0f;
            #pragma unroll
            for (int w = 0; w < 8; ++w) s += red[w];
            res[o] = s;
        }
        __syncthreads();
    }

    if (t == 0) {
        float m[6];
        for (int j = 0; j < 6; ++j) m[j] = res[j] + rbh[j];
        float mx = m[0];
        for (int j = 1; j < 6; ++j) mx = fmaxf(mx, m[j]);
        float p[6], psum = 0.0f;
        for (int j = 0; j < 6; ++j) { p[j] = exp_f(m[j] - mx); psum += p[j]; }
        float rps = __builtin_amdgcn_rcpf(psum);
        for (int j = 0; j < 6; ++j) p[j] = fmaxf(p[j] * rps, 1e-6f);

        float p_cont = p[0] + p[1];
        float p_unc  = p[2] + p[3] + p[5];
        float energy[2];
        for (int j = 0; j < 2; ++j) {
            float e0 = res[6 + j] + ebh[j];
            float e1 = res[8 + j] + ebh[2 + j];
            energy[j] = p_cont * e0 + p_unc * e1;
        }
        float p_cas = p[0];
        float p_trk = p[1] + p[2] + p[3] + p[5];
        float gate = sigmoid_f(energy[0] - 4.0f);
        float dir[3];
        for (int j = 0; j < 3; ++j) {
            float d0 = res[10 + j] + dbh[j];
            float d1 = res[13 + j] + dbh[3 + j];
            float d2 = res[16 + j] + dbh[6 + j];
            dir[j] = p_cas * d0 + p_trk * ((1.0f - gate) * d1 + gate * d2);
        }
        float* o = out + (size_t)b * 11;
        for (int j = 0; j < 6; ++j) o[j] = m[j];
        o[6] = energy[0]; o[7] = energy[1];
        o[8] = dir[0]; o[9] = dir[1]; o[10] = dir[2];
    }
}

extern "C" void kernel_launch(void* const* d_in, const int* in_sizes, int n_in,
                              void* d_out, int out_size, void* d_ws, size_t ws_size,
                              hipStream_t stream) {
    const float* coords = (const float*)d_in[0];
    const float* feats  = (const float*)d_in[1];
    const int*   bids   = (const int*)  d_in[2];
    const float* rW1 = (const float*)d_in[3];  const float* rb1 = (const float*)d_in[4];
    const float* rW2 = (const float*)d_in[5];  const float* rb2 = (const float*)d_in[6];
    const float* rWh = (const float*)d_in[7];  const float* rbh = (const float*)d_in[8];
    const float* eW1 = (const float*)d_in[9];  const float* eb1 = (const float*)d_in[10];
    const float* eW2 = (const float*)d_in[11]; const float* eb2 = (const float*)d_in[12];
    const float* eWh = (const float*)d_in[13]; const float* ebh = (const float*)d_in[14];
    const float* dW1 = (const float*)d_in[15]; const float* db1 = (const float*)d_in[16];
    const float* dW2 = (const float*)d_in[17]; const float* db2 = (const float*)d_in[18];
    const float* dWh = (const float*)d_in[19]; const float* dbh = (const float*)d_in[20];
    float* out = (float*)d_out;

    float* sums = (float*)d_ws;                    // [6][B][H]
    float* cnt  = sums + (size_t)NNETS * NB * NH;  // [B], contiguous after sums

    const int tot = NNETS * NB * NH + NB;
    neptune_init_kernel<<<(tot + 255) / 256, 256, 0, stream>>>(sums, tot);
    neptune_count_kernel<<<NPTS / 256, 256, 0, stream>>>(bids, cnt);
    neptune_mlp_kernel<<<NPTS / RPB, 256, LDS_BYTES, stream>>>(
        coords, feats, bids,
        rW1, rb1, rW2, rb2,
        eW1, eb1, eW2, eb2,
        dW1, db1, dW2, db2,
        sums);
    neptune_head_kernel<<<NB, 256, 0, stream>>>(sums, cnt, rWh, rbh, eWh, ebh, dWh, dbh, out);
}
